// OrbitalAttention_42374147342460
// MI455X (gfx1250) — compile-verified, matrix operand order checked by reading, not run on hardware
//
#include <hip/hip_runtime.h>
#include <hip/hip_bf16.h>

typedef __attribute__((ext_vector_type(16))) __bf16 v16bf;
typedef __attribute__((ext_vector_type(8)))  __bf16 v8bf;
typedef __attribute__((ext_vector_type(8)))  float  v8f;

#define NTOK   64
#define DMODEL 960
#define QKPAD  136   // 128 features + 8 pad (bank-shift)
#define VTPAD  72    // 64 keys + 8 pad
#define SPAD   68    // 64 + 4 pad (f32)
#define PPAD   72    // 64 keys + 8 pad
#define STAGE_BYTES 84480
#define LDS_BYTES   137728

// 16-bit WMMA A/B fragment: lane's 16 elements are K = base..base+7 and
// base+16..base+23, base = (lane&16)>>1. Two aligned 16B LDS loads + shuffle.
__device__ __forceinline__ v16bf frag16(const __bf16* p) {
  v8bf lo = *(const v8bf*)(p);
  v8bf hi = *(const v8bf*)(p + 16);
  return __builtin_shufflevector(lo, hi, 0,1,2,3,4,5,6,7,8,9,10,11,12,13,14,15);
}

// ---- stage X_l (component-major bf16) and head slices of Wq/Wk/Wv (transposed) ----
template<int L>
__device__ __forceinline__ void stage_l(const float* __restrict__ xb,
                                        const float* const* Wp, int h, int tid,
                                        __bf16* sX, __bf16* sW) {
  constexpr int MUL  = 256 >> L;
  constexpr int MPH  = MUL >> 3;
  constexpr int DL   = 2 * L + 1;
  constexpr int XPAD = MUL + 8;
  constexpr int XOFF = (L == 0) ? 0 : (L == 1) ? 256 : 640;

  constexpr int NX4 = NTOK * MUL * DL / 4;          // float4 chunks
  for (int i = tid; i < NX4; i += 256) {
    int idx = i * 4;
    int t = idx / (MUL * DL);
    int r = idx % (MUL * DL);
    float4 v = *(const float4*)(xb + t * DMODEL + XOFF + r);
    float vv[4] = {v.x, v.y, v.z, v.w};
#pragma unroll
    for (int e = 0; e < 4; ++e) {
      int rr = r + e;
      int m = rr / DL, comp = rr % DL;
      sX[(comp * NTOK + t) * XPAD + m] = (__bf16)vv[e];
    }
  }
  constexpr int NW = 3 * MUL * MPH;
  for (int i = tid; i < NW; i += 256) {
    int p = i / (MUL * MPH);
    int r = i % (MUL * MPH);
    int m = r / MPH, n = r % MPH;                    // n fastest -> coalesced read
    const float* W = Wp[p * 3 + L];
    sW[(p * MPH + n) * XPAD + m] = (__bf16)W[m * MUL + h * MPH + n];  // transposed
  }
}

// ---- per-l Q/K/V projection tiles out of LDS (fully unrolled k loop) ----
template<int L>
__device__ __forceinline__ void project_l(const __bf16* sX, const __bf16* sW,
                                          __bf16* sQ, __bf16* sK, __bf16* sVt,
                                          int wave, int lane) {
  constexpr int MUL   = 256 >> L;
  constexpr int MPH   = MUL >> 3;
  constexpr int KST   = MUL >> 5;
  constexpr int XPAD  = MUL + 8;
  constexpr int NJOBS = (L == 0) ? 24 : (L == 1) ? 36 : 60;
  constexpr int JPP   = NJOBS / 3;
  constexpr float ATT = 0.13363062095621219f;                       // 1/sqrt(56)
  constexpr float RSM = (L == 0) ? 0.0625f : (L == 1) ? 0.08838834764831845f : 0.125f;
  constexpr float RLD = (L == 0) ? 1.0f : (L == 1) ? 0.5773502691896258f : 0.4472135954999579f;

  const int l16  = lane & 15;
  const int rhi8 = (lane & 16) >> 1;

  for (int job = wave; job < NJOBS; job += 8) {
    int p = job / JPP;                 // 0=q 1=k 2=v
    int u = job % JPP;
    int comp, mt, nt;
    if (L == 0) { comp = 0; mt = u & 3; nt = u >> 2; }
    else        { comp = u >> 2; mt = u & 3; nt = 0; }

    const int  nloc   = nt * 16 + l16;
    const bool nvalid = nloc < MPH;
    const int  ncl    = nvalid ? nloc : (MPH - 1);   // clamped valid address

    const __bf16* arow = sX + (comp * NTOK + mt * 16 + l16) * XPAD + rhi8;
    const __bf16* brow = sW + (p * MPH + ncl) * XPAD + rhi8;

    v8f acc = {0.f,0.f,0.f,0.f,0.f,0.f,0.f,0.f};
#pragma unroll
    for (int kk = 0; kk < KST; ++kk) {
      v16bf af = frag16(arow + kk * 32);
      v16bf bf = frag16(brow + kk * 32);
      if (MPH < 16 && !nvalid) {                     // branch-free lane select (l=2 only)
#pragma unroll
        for (int j = 0; j < 16; ++j) bf[j] = (__bf16)0.f;
      }
      acc = __builtin_amdgcn_wmma_f32_16x16x32_bf16(false, af, false, bf,
                                                    (short)0, acc, false, false);
    }

    const float scale = RSM * ((p == 0) ? RLD * ATT : 1.0f);
    const int   fbase = (L == 0) ? 0 : (L == 1) ? 32 + comp * 16 : 80 + comp * 8;
    if (nvalid) {
      if (p == 2) {
#pragma unroll
        for (int r = 0; r < 8; ++r)
          sVt[(fbase + nloc) * VTPAD + mt * 16 + r + rhi8] = (__bf16)(acc[r] * scale);
      } else {
        __bf16* dst = (p == 0) ? sQ : sK;
#pragma unroll
        for (int r = 0; r < 8; ++r)
          dst[(mt * 16 + r + rhi8) * QKPAD + fbase + nloc] = (__bf16)(acc[r] * scale);
      }
    }
  }
}

__global__ __launch_bounds__(256)
void orb_attn_kernel(const float* __restrict__ x,
                     const float* __restrict__ wq0, const float* __restrict__ wq1, const float* __restrict__ wq2,
                     const float* __restrict__ wk0, const float* __restrict__ wk1, const float* __restrict__ wk2,
                     const float* __restrict__ wv0, const float* __restrict__ wv1, const float* __restrict__ wv2,
                     float* __restrict__ out)
{
  extern __shared__ __align__(16) char smem[];
  __bf16* sStage = (__bf16*)smem;                         // phase-1 X/W staging
  float*  sS  = (float*)smem;                             // 64 x 68 f32 logits (reuse)
  __bf16* sP  = (__bf16*)(smem + NTOK * SPAD * 4);        // 64 x 72 bf16 probs (reuse)
  __bf16* sQ  = (__bf16*)(smem + STAGE_BYTES);                        // 64 x 136
  __bf16* sK  = (__bf16*)(smem + STAGE_BYTES + NTOK * QKPAD * 2);     // 64 x 136
  __bf16* sVt = (__bf16*)(smem + STAGE_BYTES + 2 * NTOK * QKPAD * 2); // 128 x 72 (f-major)

  const int b    = blockIdx.x >> 3;
  const int h    = blockIdx.x & 7;
  const int tid  = threadIdx.x;
  const int wave = tid >> 5;
  const int lane = tid & 31;
  const int l16  = lane & 15;
  const int rhi8 = (lane & 16) >> 1;

  const float* xb = x + (size_t)b * NTOK * DMODEL;
  const float* Wp[9] = {wq0, wq1, wq2, wk0, wk1, wk2, wv0, wv1, wv2};

  // zero pad feature cols 120..127 of Q/K and pad feature rows 120..127 of Vt
  for (int i = tid; i < NTOK * 8; i += 256) {
    int r = i >> 3, c = 120 + (i & 7);
    sQ[r * QKPAD + c] = (__bf16)0.f;
    sK[r * QKPAD + c] = (__bf16)0.f;
    sVt[(120 + (i >> 6)) * VTPAD + (i & 63)] = (__bf16)0.f;
  }

  // ---------------- phase 1: staged Q/K/V projections, one l-block at a time --------
  {
    constexpr int XW0 = 1 * NTOK * (256 + 8);
    stage_l<0>(xb, Wp, h, tid, sStage, sStage + XW0);
    __syncthreads();
    project_l<0>(sStage, sStage + XW0, sQ, sK, sVt, wave, lane);
    __syncthreads();
  }
  {
    constexpr int XW1 = 3 * NTOK * (128 + 8);
    stage_l<1>(xb, Wp, h, tid, sStage, sStage + XW1);
    __syncthreads();
    project_l<1>(sStage, sStage + XW1, sQ, sK, sVt, wave, lane);
    __syncthreads();
  }
  {
    constexpr int XW2 = 5 * NTOK * (64 + 8);
    stage_l<2>(xb, Wp, h, tid, sStage, sStage + XW2);
    __syncthreads();
    project_l<2>(sStage, sStage + XW2, sQ, sK, sVt, wave, lane);
    __syncthreads();
  }

  // ---------------- phase 2: logits S = Q · K^T (64x64, K-dim 128) ------------------
  for (int t = wave * 2; t < wave * 2 + 2; ++t) {
    int mt = t >> 2, nt = t & 3;
    const __bf16* arow = sQ + (mt * 16 + l16) * QKPAD + rhi8;
    const __bf16* brow = sK + (nt * 16 + l16) * QKPAD + rhi8;
    v8f acc = {0.f,0.f,0.f,0.f,0.f,0.f,0.f,0.f};
#pragma unroll
    for (int kk = 0; kk < 4; ++kk) {
      v16bf af = frag16(arow + kk * 32);
      v16bf bf = frag16(brow + kk * 32);
      acc = __builtin_amdgcn_wmma_f32_16x16x32_bf16(false, af, false, bf,
                                                    (short)0, acc, false, false);
    }
#pragma unroll
    for (int r = 0; r < 8; ++r)
      sS[(mt * 16 + r + rhi8) * SPAD + nt * 16 + l16] = acc[r];
  }
  __syncthreads();

  // ---------------- phase 3: softmax over keys (f32) --------------------------------
  if (tid < NTOK) {
    float mx = -3.402823466e38f;
    for (int c = 0; c < NTOK; ++c) mx = fmaxf(mx, sS[tid * SPAD + c]);
    float s = 0.f;
    for (int c = 0; c < NTOK; ++c) {
      float e = __expf(sS[tid * SPAD + c] - mx);
      s += e;
      sS[tid * SPAD + c] = e;
    }
    float inv = 1.f / s;
    for (int c = 0; c < NTOK; ++c)
      sP[tid * PPAD + c] = (__bf16)(sS[tid * SPAD + c] * inv);
  }
  __syncthreads();

  // ---------------- phase 4: O = P · V, un-permute, residual store -------------------
  for (int t = wave * 4; t < wave * 4 + 4; ++t) {
    int mt = t >> 3, nt = t & 7;
    const __bf16* arow = sP  + (mt * 16 + l16) * PPAD  + rhi8;
    const __bf16* brow = sVt + (nt * 16 + l16) * VTPAD + rhi8;
    v8f acc = {0.f,0.f,0.f,0.f,0.f,0.f,0.f,0.f};
#pragma unroll
    for (int kk = 0; kk < 2; ++kk) {
      v16bf af = frag16(arow + kk * 32);
      v16bf bf = frag16(brow + kk * 32);
      acc = __builtin_amdgcn_wmma_f32_16x16x32_bf16(false, af, false, bf,
                                                    (short)0, acc, false, false);
    }

    int f = nt * 16 + l16;
    int go = -1;
    if (f < 32)       go = h * 32 + f;                                               // l=0
    else if (f < 80)  { int u = f - 32; go = 256 + (h * 16 + (u & 15)) * 3 + (u >> 4); } // l=1
    else if (f < 120) { int u = f - 80; go = 640 + (h *  8 + (u &  7)) * 5 + (u >> 3); } // l=2
    if (go >= 0) {
#pragma unroll
      for (int r = 0; r < 8; ++r) {
        int tok = mt * 16 + r + rhi8;
        out[(size_t)(b * NTOK + tok) * DMODEL + go] = xb[tok * DMODEL + go] + acc[r];
      }
    }
  }
}

extern "C" void kernel_launch(void* const* d_in, const int* in_sizes, int n_in,
                              void* d_out, int out_size, void* d_ws, size_t ws_size,
                              hipStream_t stream) {
  const float* x   = (const float*)d_in[0];
  const float* wq0 = (const float*)d_in[1];
  const float* wq1 = (const float*)d_in[2];
  const float* wq2 = (const float*)d_in[3];
  const float* wk0 = (const float*)d_in[4];
  const float* wk1 = (const float*)d_in[5];
  const float* wk2 = (const float*)d_in[6];
  const float* wv0 = (const float*)d_in[7];
  const float* wv1 = (const float*)d_in[8];
  const float* wv2 = (const float*)d_in[9];
  float* out = (float*)d_out;

  const int B = in_sizes[0] / (NTOK * DMODEL);   // 1024
  dim3 grid(B * 8), block(256);
  orb_attn_kernel<<<grid, block, LDS_BYTES, stream>>>(x, wq0, wq1, wq2,
                                                      wk0, wk1, wk2,
                                                      wv0, wv1, wv2, out);
}
